// RatingPredictorGNN_35064113004962
// MI455X (gfx1250) — matrix operand order repaired and checked by reading, not compiled, not run on hardware
//
#include <hip/hip_runtime.h>
#include <hip/hip_bf16.h>

typedef __attribute__((ext_vector_type(16))) _Float16 v16h;
typedef __attribute__((ext_vector_type(8)))  float    v8f;

#define NUM_USERS 100000
#define NUM_ITEMS 50000
#define N_NODES   150000
#define N_EDGES   4000000
#define HDIM      64
#define BATCH     100000

// ---------------------------------------------------------------------------
// zero a float buffer (float4 granularity; counts are multiples of 4)
__global__ void zero_f4(float4* __restrict__ p, int n4) {
    int i = blockIdx.x * blockDim.x + threadIdx.x;
    if (i < n4) p[i] = make_float4(0.f, 0.f, 0.f, 0.f);
}

// concat user_emb / item_emb into x0 (float4 granularity)
__global__ void concat_emb(const float4* __restrict__ ue, int na4,
                           const float4* __restrict__ ie, int nb4,
                           float4* __restrict__ o) {
    int i = blockIdx.x * blockDim.x + threadIdx.x;
    if (i < na4)            o[i] = ue[i];
    else if (i < na4 + nb4) o[i] = ie[i - na4];
}

// degree count (dst is fixed across layers; computed once)
__global__ void count_deg(const int* __restrict__ dst, float* __restrict__ cnt, int nE) {
    int e = blockIdx.x * blockDim.x + threadIdx.x;
    if (e < nE) atomicAdd(&cnt[dst[e]], 1.0f);
}

// Pack Wl/Wr (3 layers, [64][64] row-major, W[k][n]) into wave32 WMMA
// B-fragment layout for v_wmma_*_16x16x32: lane L, half h -> K = kc*32 + 8*(L/16) + (h&7) + 16*(h>>3),
// N = nt*16 + (L%16). Fragment index fr = kc*4 + nt.
// P layout: [layer][mat(Wl=0,Wr=1)][fr][lane][h]  (halves)
__global__ void pack_weights(const float* __restrict__ Wl, const float* __restrict__ Wr,
                             _Float16* __restrict__ P) {
    int t = blockIdx.x * blockDim.x + threadIdx.x;   // 0 .. 3*2*4096-1
    if (t >= 3 * 2 * 4096) return;
    int idx  = t & 4095;
    int mat  = (t >> 12) & 1;
    int l    = t >> 13;
    int h    = idx & 15;
    int lane = (idx >> 4) & 31;
    int fr   = idx >> 9;           // 0..7
    int kc   = fr >> 2;
    int nt   = fr & 3;
    int K = kc * 32 + 8 * (lane >> 4) + (h & 7) + 16 * (h >> 3);
    int N = nt * 16 + (lane & 15);
    const float* W = (mat ? Wr : Wl) + (size_t)l * HDIM * HDIM;
    P[t] = (_Float16)W[K * HDIM + N];
}

// scatter-add: summed[dst] += x[src]; 16 threads per edge, float4 chunks.
// x and summed are both L2-resident (38.4 MB each in 192 MB L2).
__global__ void scatter_add(const int* __restrict__ src, const int* __restrict__ dst,
                            const float* __restrict__ x, float* __restrict__ summed,
                            unsigned nWork) {
    unsigned gid = blockIdx.x * blockDim.x + threadIdx.x;
    if (gid >= nWork) return;
    int e = gid >> 4;
    int c = (gid & 15) * 4;
    int s = src[e], d = dst[e];
    float4 v = *(const float4*)(x + (size_t)s * HDIM + c);
    float* p = summed + (size_t)d * HDIM + c;
    atomicAdd(p + 0, v.x);
    atomicAdd(p + 1, v.y);
    atomicAdd(p + 2, v.z);
    atomicAdd(p + 3, v.w);
}

// Fused SAGE transform: out = relu(mean @ Wl + bl + x @ Wr)
// One wave per 16-node tile. sum_out holds `summed` on entry and is
// overwritten with the layer output (reads of a tile's rows complete before
// that tile's writes; tiles are disjoint).
__global__ __launch_bounds__(256) void sage_transform(
    const float*    __restrict__ xin,      // [N,64] layer input
    float*          __restrict__ sum_out,  // [N,64] summed -> layer output (aliased)
    const float*    __restrict__ cnt,      // [N]
    const _Float16* __restrict__ pWl,      // packed 4096 halves
    const _Float16* __restrict__ pWr,      // packed 4096 halves
    const float*    __restrict__ bias,     // [64]
    int nTiles)
{
    int wave = threadIdx.x >> 5;
    int lane = threadIdx.x & 31;
    int tile = blockIdx.x * (blockDim.x >> 5) + wave;
    if (tile >= nTiles) return;            // wave-uniform: EXEC all-1 for WMMA

    int row0 = tile * 16;
    int m    = lane & 15;                  // A-row / C-col index
    int g    = lane >> 4;                  // lane half-group
    int row  = row0 + m;

    float inv = 1.0f / fmaxf(cnt[row], 1.0f);

    const float* xrow = xin     + (size_t)row * HDIM;
    const float* srow = sum_out + (size_t)row * HDIM;

    // A fragments (16-bit A 16x32 layout): halves 0..7 -> K = kc*32 + 8g + h,
    // halves 8..15 -> K = kc*32 + 16 + 8g + (h-8)
    v16h aM[2], aX[2];
#pragma unroll
    for (int kc = 0; kc < 2; ++kc) {
        int kbase = kc * 32 + 8 * g;
#pragma unroll
        for (int h = 0; h < 8; ++h) {
            int kLo = kbase + h;
            int kHi = kbase + 16 + h;
            aX[kc][h]     = (_Float16)xrow[kLo];
            aX[kc][8 + h] = (_Float16)xrow[kHi];
            aM[kc][h]     = (_Float16)(srow[kLo] * inv);
            aM[kc][8 + h] = (_Float16)(srow[kHi] * inv);
        }
    }

#pragma unroll
    for (int nt = 0; nt < 4; ++nt) {
        // accumulator init with bias: D[M][N], N = nt*16 + m for every VGPR
        float bv = bias[nt * 16 + m];
        v8f acc = {bv, bv, bv, bv, bv, bv, bv, bv};
#pragma unroll
        for (int kc = 0; kc < 2; ++kc) {
            v16h bL = *(const v16h*)(pWl + (((kc * 4 + nt) * 32 + lane) << 4));
            acc = __builtin_amdgcn_wmma_f32_16x16x32_f16(
                false, aM[kc], false, bL, (short)0, acc, false, false);
            v16h bR = *(const v16h*)(pWr + (((kc * 4 + nt) * 32 + lane) << 4));
            acc = __builtin_amdgcn_wmma_f32_16x16x32_f16(
                false, aX[kc], false, bR, (short)0, acc, false, false);
        }
        // relu + store; D layout: VGPR v -> row v (lanes 0-15) / v+8 (lanes 16-31)
#pragma unroll
        for (int v = 0; v < 8; ++v) {
            float val = fmaxf(acc[v], 0.0f);
            int M = v + 8 * g;
            sum_out[(size_t)(row0 + M) * HDIM + nt * 16 + m] = val;
        }
    }
}

// Final MLP: rating = clip(relu(pair @ W1 + b1) @ W2 + b2, 1, 5)
__global__ __launch_bounds__(256) void mlp_pred(
    const float* __restrict__ xf,
    const int* __restrict__ uid, const int* __restrict__ iid,
    const float* __restrict__ W1, const float* __restrict__ b1,
    const float* __restrict__ W2, const float* __restrict__ b2,
    float* __restrict__ out, int nB)
{
    __shared__ float sW1[128 * 32];
    __shared__ float sB1[32];
    __shared__ float sW2[32];
    for (int i = threadIdx.x; i < 128 * 32; i += blockDim.x) sW1[i] = W1[i];
    if (threadIdx.x < 32) {
        sB1[threadIdx.x] = b1[threadIdx.x];
        sW2[threadIdx.x] = W2[threadIdx.x];
    }
    __syncthreads();

    int b = blockIdx.x * blockDim.x + threadIdx.x;
    if (b >= nB) return;

    const float* ue = xf + (size_t)uid[b] * HDIM;
    const float* ie = xf + (size_t)(iid[b] + NUM_USERS) * HDIM;

    float acc[32];
#pragma unroll
    for (int j = 0; j < 32; ++j) acc[j] = sB1[j];
    for (int k = 0; k < HDIM; ++k) {
        float p = ue[k];
        const float* w = sW1 + k * 32;
#pragma unroll
        for (int j = 0; j < 32; ++j) acc[j] += p * w[j];
    }
    for (int k = 0; k < HDIM; ++k) {
        float p = ie[k];
        const float* w = sW1 + (HDIM + k) * 32;
#pragma unroll
        for (int j = 0; j < 32; ++j) acc[j] += p * w[j];
    }
    float r = b2[0];
#pragma unroll
    for (int j = 0; j < 32; ++j) r += fmaxf(acc[j], 0.0f) * sW2[j];
    out[b] = fminf(fmaxf(r, 1.0f), 5.0f);
}

// ---------------------------------------------------------------------------
extern "C" void kernel_launch(void* const* d_in, const int* in_sizes, int n_in,
                              void* d_out, int out_size, void* d_ws, size_t ws_size,
                              hipStream_t stream) {
    const int*   edge     = (const int*)d_in[0];   // [2, E]
    const int*   uid      = (const int*)d_in[1];
    const int*   iid      = (const int*)d_in[2];
    const float* user_emb = (const float*)d_in[3];
    const float* item_emb = (const float*)d_in[4];
    const float* Wl       = (const float*)d_in[5]; // [3,64,64]
    const float* bl       = (const float*)d_in[6]; // [3,64]
    const float* Wr       = (const float*)d_in[7]; // [3,64,64]
    const float* W1       = (const float*)d_in[8]; // [128,32]
    const float* b1       = (const float*)d_in[9];
    const float* W2       = (const float*)d_in[10];
    const float* b2       = (const float*)d_in[11];
    float*       out      = (float*)d_out;

    const int* src = edge;
    const int* dst = edge + N_EDGES;

    // workspace carve-up (~78 MB): two ping-pong feature buffers, degree, packed W
    const size_t nFeat = (size_t)N_NODES * HDIM;          // 9.6M floats
    float*     bufA   = (float*)d_ws;
    float*     bufB   = bufA + nFeat;
    float*     cnt    = bufB + nFeat;
    _Float16*  packed = (_Float16*)(cnt + N_NODES);       // 3*2*4096 halves

    const int TPB = 256;

    // x0 = concat(user_emb, item_emb)
    {
        int na4 = NUM_USERS * HDIM / 4, nb4 = NUM_ITEMS * HDIM / 4;
        int tot = na4 + nb4;
        concat_emb<<<(tot + TPB - 1) / TPB, TPB, 0, stream>>>(
            (const float4*)user_emb, na4, (const float4*)item_emb, nb4, (float4*)bufA);
    }
    // degree counts (once)
    zero_f4<<<(N_NODES / 4 + TPB - 1) / TPB, TPB, 0, stream>>>((float4*)cnt, N_NODES / 4);
    count_deg<<<(N_EDGES + TPB - 1) / TPB, TPB, 0, stream>>>(dst, cnt, N_EDGES);
    // pack all layer weights into WMMA B-fragment layout (once)
    pack_weights<<<(3 * 2 * 4096 + TPB - 1) / TPB, TPB, 0, stream>>>(Wl, Wr, packed);

    const int nTiles = N_NODES / 16;                      // 9375
    const int wavesPerBlock = TPB / 32;
    float* xin = bufA;
    float* xo  = bufB;
    for (int l = 0; l < 3; ++l) {
        zero_f4<<<((int)(nFeat / 4) + TPB - 1) / TPB, TPB, 0, stream>>>(
            (float4*)xo, (int)(nFeat / 4));
        unsigned nWork = (unsigned)N_EDGES * 16u;
        scatter_add<<<(nWork + TPB - 1) / TPB, TPB, 0, stream>>>(src, dst, xin, xo, nWork);
        sage_transform<<<(nTiles + wavesPerBlock - 1) / wavesPerBlock, TPB, 0, stream>>>(
            xin, xo, cnt,
            packed + (size_t)l * 8192, packed + (size_t)l * 8192 + 4096,
            bl + l * HDIM, nTiles);
        float* t = xin; xin = xo; xo = t;
    }

    mlp_pred<<<(BATCH + TPB - 1) / TPB, TPB, 0, stream>>>(
        xin, uid, iid, W1, b1, W2, b2, out, BATCH);
}
